// FBPinn_71141838291395
// MI455X (gfx1250) — matrix operand order repaired and checked by reading, not compiled, not run on hardware
//
#include <hip/hip_runtime.h>
#include <cmath>

// ---------------------------------------------------------------------------
// FBPinn forward for MI455X (gfx1250, wave32).
// Layer 2 (32x32) on V_WMMA_F32_16X16X32_F16 with W2^T as the A matrix
// (weight-stationary, loaded once per window) and the h1 activations as the
// B matrix (N = point column = lane&15, built per-lane with no cross-lane
// traffic). D holds h2 point-per-lane, so layer 3 is a per-lane dot; the only
// cross-lane op is ONE half-swap shuffle per 16-point group at kernel end.
// Layer 1 uses native packed f16 (v_pk_fma_f16 + v_tanh_f16) if the f16 tanh
// builtin exists under either known name; otherwise a clean f32 path
// (f32 fma -> v_tanh_f32 -> paired v_cvt_pk_f16_f32).
// ---------------------------------------------------------------------------

typedef __attribute__((ext_vector_type(16))) _Float16 v16h;
typedef __attribute__((ext_vector_type(2)))  _Float16 v2h;
typedef __attribute__((ext_vector_type(8)))  float    v8f;

#define NWIN   30
#define NEU    32
#define GROUPS 8                    // 16-point groups per wave
#define WAVES  8                    // waves per block (256 threads = 8 wave32)
#define BLOCK  (WAVES * 32)
#define PTS_PER_BLOCK (WAVES * GROUPS * 16)

#define LOG2E 1.4426950408889634f

__device__ __forceinline__ float fexp2(float t) {
#if __has_builtin(__builtin_amdgcn_exp2f)
    return __builtin_amdgcn_exp2f(t);
#else
    return exp2f(t);
#endif
}
__device__ __forceinline__ float frcp(float t) {
#if __has_builtin(__builtin_amdgcn_rcpf)
    return __builtin_amdgcn_rcpf(t);
#else
    return 1.0f / t;
#endif
}
__device__ __forceinline__ float ftanh(float v) {
#if __has_builtin(__builtin_amdgcn_tanhf)
    return __builtin_amdgcn_tanhf(v);
#else
    float e = fexp2(v * (2.0f * LOG2E));     // e^(2v)
    return 1.0f - 2.0f * frcp(e + 1.0f);     // saturates via inf->rcp->0
#endif
}
__device__ __forceinline__ float fsigmoid(float t) {
    return frcp(1.0f + fexp2(-t * LOG2E));
}

// Native f16 tanh: probe both plausible builtin names (f32 one is "tanhf",
// AMDGPU f16 trans builtins conventionally use the "h" suffix).
#if __has_builtin(__builtin_amdgcn_tanh_f16)
#define HAVE_TANH16 1
__device__ __forceinline__ _Float16 ftanh16(_Float16 v) {
    return __builtin_amdgcn_tanh_f16(v);
}
#elif __has_builtin(__builtin_amdgcn_tanhh)
#define HAVE_TANH16 1
__device__ __forceinline__ _Float16 ftanh16(_Float16 v) {
    return __builtin_amdgcn_tanhh(v);
}
#else
#define HAVE_TANH16 0
#endif

__global__ void __launch_bounds__(BLOCK)
fbpinn_wmma_kernel(const float* __restrict__ x,
                   const float* __restrict__ W1, const float* __restrict__ B1,
                   const float* __restrict__ W2, const float* __restrict__ B2,
                   const float* __restrict__ W3, const float* __restrict__ B3,
                   float* __restrict__ out, int n)
{
    const int lane     = threadIdx.x & 31;
    const int wave     = threadIdx.x >> 5;
    const int p        = lane & 15;          // this lane's point within a group
    const bool lowHalf = (lane < 16);
    const int base     = blockIdx.x * PTS_PER_BLOCK + wave * (GROUPS * 16);

    // A-fragment K bases (this code's contiguous k order, same as B):
    //   lanes 0-15 : K 0..7 in v0-3, K 8..15 in v4-7
    //   lanes 16-31: K 16..23 in v0-3, K 24..31 in v4-7
    const int ka0 = lowHalf ? 0 : 16;
    const int ka1 = lowHalf ? 8 : 24;
    // B-fragment K base (contiguous order, as A: lanes 0-15 rows K0..15, lanes 16-31 K16..31)
    const int kb  = lowHalf ? 0 : 16;
    // C/D row base per tile (C/D 16x16: lanes 0-15 -> M=v, lanes 16-31 -> M=v+8)
    const int mr0 = lowHalf ? 0 : 8;     // tile0 rows = output neurons 0..15
    const int mr1 = lowHalf ? 16 : 24;   // tile1 rows = output neurons 16..31

    float xs[GROUPS], acc[GROUPS];
    int   idx[GROUPS];
#pragma unroll
    for (int g = 0; g < GROUPS; ++g) {
        int i  = base + g * 16 + p;
        idx[g] = i;
        int ic = (i < n) ? i : (n - 1);
        xs[g]  = x[ic];
        acc[g] = 0.0f;                   // per-lane HALF of the prediction
    }

    const float width = 100.0f / (float)NWIN;

    for (int w = 0; w < NWIN; ++w) {
        // ---- window geometry (matches _partition(); overlap midpoints = uniform grid)
        float sub_lo = (w == 0)        ? 0.0f   : ((float)w - 0.125f) * width;
        float sub_hi = (w == NWIN - 1) ? 100.0f : ((float)w + 1.125f) * width;
        float m    = 0.5f * (sub_lo + sub_hi);
        float sinv = frcp(0.5f * (sub_hi - sub_lo));
        float lmid = (float)w * width;
        float rmid = (float)(w + 1) * width;

        const float* w1  = W1 + w * NEU;        // [1,32]
        const float* bb1 = B1 + w * NEU;
        const float* w2  = W2 + w * NEU * NEU;  // [32,32] (k_in, n_out) row-major
        const float* bb2 = B2 + w * NEU;
        const float* w3  = W3 + w * NEU;        // [32,1]
        const float  b3h = 0.5f * B3[w];        // half bias per 16-lane half

        // ---- layer-1 params at this lane's B-fragment K rows (kb..kb+15)
#if HAVE_TANH16
        v2h w1h[8], b1h[8];
#pragma unroll
        for (int t = 0; t < 8; ++t) {
            w1h[t] = (v2h){(_Float16)w1[kb + 2*t], (_Float16)w1[kb + 2*t + 1]};
            b1h[t] = (v2h){(_Float16)bb1[kb + 2*t], (_Float16)bb1[kb + 2*t + 1]};
        }
#else
        float w1v[16], b1v[16];
#pragma unroll
        for (int t = 0; t < 16; ++t) { w1v[t] = w1[kb + t]; b1v[t] = bb1[kb + t]; }
#endif

        // ---- A fragments = W2^T tiles (rows = output neurons), f32 -> f16.
        const int m0 = p;        // tile0: output neuron row = lane&15
        const int m1 = 16 + p;   // tile1: output neuron row = 16 + (lane&15)
        v16h A0, A1;
#pragma unroll
        for (int t = 0; t < 8; ++t) {
            A0[t]     = (_Float16)w2[(ka0 + t) * NEU + m0];
            A0[8 + t] = (_Float16)w2[(ka1 + t) * NEU + m0];
            A1[t]     = (_Float16)w2[(ka0 + t) * NEU + m1];
            A1[8 + t] = (_Float16)w2[(ka1 + t) * NEU + m1];
        }

        // ---- per-lane layer-2 bias rows and layer-3 weights for this lane's D rows
        float c0[8], c1[8], w3a[8], w3b[8];
#pragma unroll
        for (int v = 0; v < 8; ++v) {
            c0[v]  = bb2[mr0 + v];   c1[v]  = bb2[mr1 + v];
            w3a[v] = w3[mr0 + v];    w3b[v] = w3[mr1 + v];
        }

#pragma unroll
        for (int g = 0; g < GROUPS; ++g) {
            float xv = xs[g];
            float xn = (xv - m) * sinv;

            // layer 1 -> f16 B fragment (K rows kb..kb+15 for this lane's point)
            v16h Bf;
#if HAVE_TANH16
            _Float16 xnh = (_Float16)xn;
            v2h xn2 = (v2h){xnh, xnh};
#pragma unroll
            for (int t = 0; t < 8; ++t) {
                v2h h = w1h[t] * xn2 + b1h[t];     // v_pk_fma_f16
                Bf[2*t]     = ftanh16(h[0]);       // v_tanh_f16
                Bf[2*t + 1] = ftanh16(h[1]);
            }
#else
#pragma unroll
            for (int t = 0; t < 16; ++t)
                Bf[t] = (_Float16)ftanh(xn * w1v[t] + b1v[t]);  // fma+tanh f32, cvt_pk pairs
#endif

            // layer 2: two WMMAs (output-neuron tiles), C pre-loaded with b2 rows
            v8f C0, C1;
#pragma unroll
            for (int v = 0; v < 8; ++v) { C0[v] = c0[v]; C1[v] = c1[v]; }
            v8f D0 = __builtin_amdgcn_wmma_f32_16x16x32_f16(
                         false, A0, false, Bf, (short)0, C0, false, false);
            v8f D1 = __builtin_amdgcn_wmma_f32_16x16x32_f16(
                         false, A1, false, Bf, (short)0, C1, false, false);

            // layer 3: per-lane dot over this lane's 16 h2 values (half the sum)
            float part = b3h;
#pragma unroll
            for (int v = 0; v < 8; ++v)
                part += ftanh(D0[v]) * w3a[v] + ftanh(D1[v]) * w3b[v];

            // window function + masked accumulate (identical in both halves)
            float win = fsigmoid(xv - lmid) * fsigmoid(rmid - xv);   // SIGMA = 1
            acc[g] += (win > 0.001f) ? (win * part) : 0.0f;
        }
    }

    // combine the two 16-lane halves (one shuffle per group), apply ansatz, store
#pragma unroll
    for (int g = 0; g < GROUPS; ++g) {
        float tot = acc[g] + __shfl_xor(acc[g], 16, 32);
        if (lowHalf && idx[g] < n)
            out[idx[g]] = ftanh(xs[g]) * tot;
    }
}

extern "C" void kernel_launch(void* const* d_in, const int* in_sizes, int n_in,
                              void* d_out, int out_size, void* d_ws, size_t ws_size,
                              hipStream_t stream)
{
    const float* x  = (const float*)d_in[0];
    const float* W1 = (const float*)d_in[1];
    const float* b1 = (const float*)d_in[2];
    const float* W2 = (const float*)d_in[3];
    const float* b2 = (const float*)d_in[4];
    const float* W3 = (const float*)d_in[5];
    const float* b3 = (const float*)d_in[6];
    float* out = (float*)d_out;
    int n = in_sizes[0];
    int grid = (n + PTS_PER_BLOCK - 1) / PTS_PER_BLOCK;
    if (grid < 1) grid = 1;
    fbpinn_wmma_kernel<<<grid, BLOCK, 0, stream>>>(x, W1, b1, W2, b2, W3, b3, out, n);
}
